// CausalSelfAttention_18519898980524
// MI455X (gfx1250) — compile-verified, matrix operand order checked by reading, not run on hardware
//
#include <hip/hip_runtime.h>
#include <hip/hip_bf16.h>

// ---------------------------------------------------------------------------
// CausalSelfAttention (B=4, S=2048, E=768, H=12, D=64) with ALiBi, fp32 I/O.
// bf16 WMMA for all 4 matmuls, fp32 accumulation, fp32 online softmax.
// GEMM uses the CDNA5 Tensor Data Mover (tensor_load_to_lds) double-buffered:
// wave0 DMAs the next K-tile (with hardware LDS padding matching the WMMA
// fragment layout) while all waves run v_wmma on the current tile.
// ---------------------------------------------------------------------------

typedef __attribute__((ext_vector_type(16))) __bf16 v16bf;
typedef __attribute__((ext_vector_type(8)))  float  v8f;

#define H_HEADS 12
#define DHEAD   64

#if __has_builtin(__builtin_amdgcn_tensor_load_to_lds) && \
    __has_builtin(__builtin_amdgcn_s_wait_tensorcnt)
#define USE_TDM 1
#else
#define USE_TDM 0
#endif

__device__ __forceinline__ __bf16 f2bf(float f) {
  unsigned u = __builtin_bit_cast(unsigned, f);
  unsigned r = u + 0x7FFFu + ((u >> 16) & 1u);       // round-to-nearest-even
  unsigned short hs = (unsigned short)(r >> 16);
  return __builtin_bit_cast(__bf16, hs);
}

// packed pair: low 16 = rne(a), high 16 = rne(b)
__device__ __forceinline__ unsigned pk_bf16(float a, float b) {
#if __has_builtin(__builtin_amdgcn_cvt_pk_bf16_f32)
  auto p = __builtin_amdgcn_cvt_pk_bf16_f32(a, b);
  return __builtin_bit_cast(unsigned, p);
#else
  unsigned ua = __builtin_bit_cast(unsigned, a);
  unsigned ub = __builtin_bit_cast(unsigned, b);
  ua = ua + 0x7FFFu + ((ua >> 16) & 1u);
  ub = ub + 0x7FFFu + ((ub >> 16) & 1u);
  return (ub & 0xFFFF0000u) | (ua >> 16);
#endif
}

union Frag16 { v16bf v; uint4 u[2]; };

__device__ __forceinline__ v8f v8f_zero() {
  v8f z = {0.f, 0.f, 0.f, 0.f, 0.f, 0.f, 0.f, 0.f};
  return z;
}

#if USE_TDM
typedef unsigned u32x4 __attribute__((ext_vector_type(4)));
typedef int      i32x8 __attribute__((ext_vector_type(8)));
typedef int      i32x4 __attribute__((ext_vector_type(4)));

// TDM: DMA a [tileRows x 32-half] 2D tile (row stride = K halves) from global
// bf16 into LDS at ldsByteOff, inserting 16B of LDS padding after every 64B
// row => LDS row stride = 80B, exactly the padded WMMA-fragment layout.
__device__ __forceinline__ void tdm_load_tile(unsigned ldsByteOff,
                                              const __bf16* gptr,
                                              int tileRows, int K, int tensorRows) {
  unsigned long long ga = (unsigned long long)(uintptr_t)(const void*)gptr;
  u32x4 g0;
  g0[0] = 1u;                                            // count=1 (valid, user)
  g0[1] = ldsByteOff;                                    // lds_addr (bytes)
  g0[2] = (unsigned)ga;                                  // global_addr[31:0]
  g0[3] = (unsigned)((ga >> 32) & 0x1FFFFFFu) | (2u << 30); // addr[56:32]|type=2
  i32x8 g1;
  g1[0] = (int)((1u << 16)        // data_size = 2 bytes
              | (1u << 20)        // pad_enable
              | (3u << 22)        // pad_interval: 16 DWORDs (64B) between pads
              | (3u << 25));      // pad_amount:   4 DWORDs (16B)
  g1[1] = (int)(((unsigned)K & 0xFFFFu) << 16);          // tensor_dim0 lo16
  g1[2] = (int)((((unsigned)K >> 16) & 0xFFFFu)          // tensor_dim0 hi16
              | (((unsigned)tensorRows & 0xFFFFu) << 16)); // tensor_dim1 lo16
  g1[3] = (int)(((unsigned)tensorRows >> 16)             // tensor_dim1 hi16
              | (32u << 16));                            // tile_dim0 = 32 elems
  g1[4] = (int)((unsigned)tileRows);                     // tile_dim1; tile_dim2=0
  g1[5] = K;                                             // tensor_dim0_stride lo32
  g1[6] = 0;                                             // stride hi16 | dim1_stride
  g1[7] = 0;
  i32x4 z4 = {0, 0, 0, 0};
  i32x8 z8 = {0, 0, 0, 0, 0, 0, 0, 0};
  // amdgpu-toolchain (clang-23) 6-arg form: (g0, g1, g2, g3, g4, cpol)
  __builtin_amdgcn_tensor_load_to_lds(g0, g1, z4, z4, z8, 0);
}
#endif

// ---------------------------------------------------------------------------
// One-time: fp32 -> bf16, 8 elements / thread, packed conversions.
// ---------------------------------------------------------------------------
__global__ __launch_bounds__(256) void cvt_x(const float* __restrict__ in,
                                             __bf16* __restrict__ out, long n) {
  long i = ((long)blockIdx.x * 256 + threadIdx.x) * 8;
  if (i >= n) return;
  float4 a = *(const float4*)(in + i);
  float4 b = *(const float4*)(in + i + 4);
  uint4 o;
  o.x = pk_bf16(a.x, a.y); o.y = pk_bf16(a.z, a.w);
  o.z = pk_bf16(b.x, b.y); o.w = pk_bf16(b.z, b.w);
  *(uint4*)(out + i) = o;
}

// ---------------------------------------------------------------------------
// One-time: W fp32 [K,N] row-major -> Wt bf16 [N,K] (transposed), LDS tile.
// ---------------------------------------------------------------------------
__global__ __launch_bounds__(256) void transpose_w(const float* __restrict__ W,
                                                   __bf16* __restrict__ Wt,
                                                   int N, int K) {
  __shared__ __bf16 tile[64 * 72];
  const int t  = threadIdx.x;
  const int k0 = blockIdx.x * 64;
  const int n0 = blockIdx.y * 64;
#pragma unroll
  for (int i = 0; i < 4; ++i) {
    int q  = t + 256 * i;
    int c  = q >> 4;                   // k-local 0..63
    int nn = (q & 15) << 2;            // n-local 0..60 step 4
    float4 fv = *(const float4*)(W + (long)(k0 + c) * N + n0 + nn);
    tile[(nn + 0) * 72 + c] = f2bf(fv.x);
    tile[(nn + 1) * 72 + c] = f2bf(fv.y);
    tile[(nn + 2) * 72 + c] = f2bf(fv.z);
    tile[(nn + 3) * 72 + c] = f2bf(fv.w);
  }
  __syncthreads();
#pragma unroll
  for (int i = 0; i < 2; ++i) {
    int q  = t + 256 * i;
    int rl = q >> 3;                   // n-local 0..63
    int kc = (q & 7) << 3;             // k-local 0..56 step 8
    *(uint4*)(Wt + (long)(n0 + rl) * K + k0 + kc) = *(uint4*)&tile[rl * 72 + kc];
  }
}

// ---------------------------------------------------------------------------
// GEMM: C[M,N] = A[M,K](bf16 rm) * Wt[N,K](bf16 pre-transposed) + bias.
// mode 0: fp32 row-major outF.   mode 1: bf16 [B,H,S,D] head-split outQKV.
// Block tile 128x128, 256 threads (8 waves), wave tile 64x32 (4x2 frags).
// TDM path: double-buffered tensor_load_to_lds issued by wave 0, overlapped
// with WMMA compute; fallback: batched global->LDS copies.
// ---------------------------------------------------------------------------
#define BM 128
#define BN 128
#define BK 32
#define LDT (BK + 8)            // 40 halves / row -> 80B stride (TDM pad)
#define STG ((BM + BN) * LDT)   // halves per stage

__global__ __launch_bounds__(256) void gemm_kernel(
    const __bf16* __restrict__ A, const __bf16* __restrict__ Wt,
    const float* __restrict__ bias, float* __restrict__ outF,
    __bf16* __restrict__ outQKV, int M, int N, int K, int S, int mode)
{
  __shared__ __bf16 lds[2 * STG];     // 2 x (10KB A + 10KB B)

  const int t    = threadIdx.x;
  const int m0   = blockIdx.y * BM;
  const int n0   = blockIdx.x * BN;
  const int wave = t >> 5, lane = t & 31, sel = lane >> 4, r = lane & 15;
  const int wm   = (wave >> 2) * 64;     // 0 / 64
  const int wn   = (wave & 3) * 32;      // 0 / 32 / 64 / 96
  const int nIter = K / BK;

  v8f acc[4][2];
#pragma unroll
  for (int i = 0; i < 4; ++i)
#pragma unroll
    for (int j = 0; j < 2; ++j) acc[i][j] = v8f_zero();

#if USE_TDM
  const unsigned ldsBase = (unsigned)(unsigned long long)(uintptr_t)(void*)&lds[0];
  if (wave == 0) {
    tdm_load_tile(ldsBase,                A + (long)m0 * K, BM, K, M);
    tdm_load_tile(ldsBase + BM * LDT * 2, Wt + (long)n0 * K, BN, K, N);
    __builtin_amdgcn_s_wait_tensorcnt(0);
  }
  __syncthreads();
#endif

  for (int it = 0; it < nIter; ++it) {
    const int k0 = it * BK;
#if USE_TDM
    const int stage = it & 1;
    if (wave == 0 && it + 1 < nIter) {      // DMA next tile into other buffer
      const unsigned off = ldsBase + (unsigned)((1 - stage) * STG) * 2u;
      tdm_load_tile(off,                A + (long)m0 * K + k0 + BK, BM, K, M);
      tdm_load_tile(off + BM * LDT * 2, Wt + (long)n0 * K + k0 + BK, BN, K, N);
    }
    const __bf16* aB = &lds[stage * STG];
    const __bf16* bB = aB + BM * LDT;
#else
    // fallback: batched copies (all loads issued before stores)
    const __bf16* aB = &lds[0];
    const __bf16* bB = aB + BM * LDT;
    {
      const int srow = t >> 2;              // 0..63
      const int skc  = (t & 3) << 3;        // 0..24 step 8
      uint4 a0 = *(const uint4*)(A  + (long)(m0 + srow)      * K + k0 + skc);
      uint4 a1 = *(const uint4*)(A  + (long)(m0 + srow + 64) * K + k0 + skc);
      uint4 b0 = *(const uint4*)(Wt + (long)(n0 + srow)      * K + k0 + skc);
      uint4 b1 = *(const uint4*)(Wt + (long)(n0 + srow + 64) * K + k0 + skc);
      *(uint4*)&lds[(srow)      * LDT + skc] = a0;
      *(uint4*)&lds[(srow + 64) * LDT + skc] = a1;
      *(uint4*)&lds[BM * LDT + (srow)      * LDT + skc] = b0;
      *(uint4*)&lds[BM * LDT + (srow + 64) * LDT + skc] = b1;
      if (k0 + BK < K)
        __builtin_prefetch(A + (long)(m0 + srow) * K + k0 + BK + skc, 0, 1);
    }
    __syncthreads();
#endif

    // ---- fragments (16-bit A/B lane layout, two b128 reads each) -------
    Frag16 af[4], bfr[2];
#pragma unroll
    for (int mi = 0; mi < 4; ++mi) {
      const __bf16* rp = &aB[(wm + mi * 16 + r) * LDT];
      af[mi].u[0] = *(const uint4*)(rp + sel * 8);
      af[mi].u[1] = *(const uint4*)(rp + 16 + sel * 8);
    }
#pragma unroll
    for (int ni = 0; ni < 2; ++ni) {
      const __bf16* rp = &bB[(wn + ni * 16 + r) * LDT];
      bfr[ni].u[0] = *(const uint4*)(rp + sel * 8);
      bfr[ni].u[1] = *(const uint4*)(rp + 16 + sel * 8);
    }
#pragma unroll
    for (int mi = 0; mi < 4; ++mi)
#pragma unroll
      for (int ni = 0; ni < 2; ++ni)
        acc[mi][ni] = __builtin_amdgcn_wmma_f32_16x16x32_bf16(
            false, af[mi].v, false, bfr[ni].v, (short)0, acc[mi][ni],
            false, false);

#if USE_TDM
    __syncthreads();                          // all waves done with this stage
    if (wave == 0) __builtin_amdgcn_s_wait_tensorcnt(0);  // next stage landed
    __syncthreads();                          // publish to all waves
#else
    __syncthreads();
#endif
  }

  // ---- epilogue: + bias, scatter per output mode -----------------------
#pragma unroll
  for (int mi = 0; mi < 4; ++mi) {
#pragma unroll
    for (int ni = 0; ni < 2; ++ni) {
      int col  = n0 + wn + ni * 16 + r;
      float bv = bias[col];
#pragma unroll
      for (int e = 0; e < 8; ++e) {
        int row  = m0 + wm + mi * 16 + sel * 8 + e;   // C-layout row mapping
        float vv = acc[mi][ni][e] + bv;
        if (mode == 0) {
          outF[(long)row * N + col] = vv;
        } else {
          int b = row / S, s = row - b * S;
          int head = col >> 6, d = col & 63;
          outQKV[(((long)(b * H_HEADS + head)) * S + s) * DHEAD + d] = f2bf(vv);
        }
      }
    }
  }
}

// ---------------------------------------------------------------------------
// Flash attention with ALiBi + causal mask.
// Grid: (S/64, B*H). Block: 128 threads = 4 waves; wave w owns 16 query rows.
// 32 keys/iter; K staged row-major (== B-frag layout for Q.K^T),
// V staged transposed (== B-frag layout for P.V). P transposed via LDS.
// ---------------------------------------------------------------------------
__global__ __launch_bounds__(128) void attn_kernel(
    const __bf16* __restrict__ Q, const __bf16* __restrict__ Kh,
    const __bf16* __restrict__ V, __bf16* __restrict__ O, int S)
{
  __shared__ __bf16 lK[32 * 72];        // 32 keys x 64 d (+8 pad)   4.5 KB
  __shared__ __bf16 lVt[64 * 40];       // 64 d x 32 keys (+8 pad)   5.0 KB
  __shared__ __bf16 lP[4 * 16 * 40];    // per-wave 16x32 P scratch  5.0 KB

  const int t    = threadIdx.x;
  const int wave = t >> 5, lane = t & 31, sel = lane >> 4, r = lane & 15;
  const int bh   = blockIdx.y;
  const int h    = bh % H_HEADS;
  const int qrow0 = blockIdx.x * 64 + wave * 16;

  const float L2E    = 1.44269504088896f;
  const float slope  = exp2f(-(2.0f / 3.0f) * (float)(h + 1)); // ALIBI_SLOPES[h]
  const float SCL    = 0.125f * L2E;                           // D^-0.5 * log2 e
  const float slope2 = slope * L2E;

  const __bf16* Qb = Q  + (long)bh * S * DHEAD;
  const __bf16* Kb = Kh + (long)bh * S * DHEAD;
  const __bf16* Vb = V  + (long)bh * S * DHEAD;

  // Q fragments for this wave's 16 rows (K-dim = d, split 0..31 / 32..63)
  Frag16 qa[2];
#pragma unroll
  for (int kf = 0; kf < 2; ++kf) {
    const __bf16* rp = Qb + (long)(qrow0 + r) * DHEAD + kf * 32;
    qa[kf].u[0] = *(const uint4*)(rp + sel * 8);
    qa[kf].u[1] = *(const uint4*)(rp + 16 + sel * 8);
  }

  v8f oacc[4];
#pragma unroll
  for (int i = 0; i < 4; ++i) oacc[i] = v8f_zero();
  float mrun[8], lrun[8];
#pragma unroll
  for (int e = 0; e < 8; ++e) { mrun[e] = -__builtin_inff(); lrun[e] = 0.0f; }

  const int krA = t >> 3;               // 0..15
  const int ccA = (t & 7) << 3;         // 0..56 step 8
  const int krB = krA + 16;             // 16..31

  const int ntiles = blockIdx.x * 2 + 2;     // causal bound for this block
  for (int it = 0; it < ntiles; ++it) {
    const int kb = it * 32;
    // ---- stage K (row-major) and V (transposed); loads batched ---------
    {
      uint4 kv0 = *(const uint4*)(Kb + (long)(kb + krA) * DHEAD + ccA);
      uint4 kv1 = *(const uint4*)(Kb + (long)(kb + krB) * DHEAD + ccA);
      uint4 vv0 = *(const uint4*)(Vb + (long)(kb + krA) * DHEAD + ccA);
      uint4 vv1 = *(const uint4*)(Vb + (long)(kb + krB) * DHEAD + ccA);
      if (it + 1 < ntiles)
        __builtin_prefetch(Kb + (long)(kb + 32 + krA) * DHEAD + ccA, 0, 1);
      *(uint4*)&lK[krA * 72 + ccA] = kv0;
      *(uint4*)&lK[krB * 72 + ccA] = kv1;
      union { uint4 u; __bf16 hh[8]; } a0, a1;
      a0.u = vv0; a1.u = vv1;
#pragma unroll
      for (int j = 0; j < 8; ++j) {
        lVt[(ccA + j) * 40 + krA] = a0.hh[j];
        lVt[(ccA + j) * 40 + krB] = a1.hh[j];
      }
    }
    __syncthreads();

    // ---- scores: S = Q . K^T  (two 16-key subtiles) --------------------
    v8f sacc[2]; sacc[0] = v8f_zero(); sacc[1] = v8f_zero();
#pragma unroll
    for (int ns = 0; ns < 2; ++ns) {
#pragma unroll
      for (int kf = 0; kf < 2; ++kf) {
        Frag16 kfr;
        const __bf16* rp = &lK[(ns * 16 + r) * 72 + kf * 32];
        kfr.u[0] = *(const uint4*)(rp + sel * 8);
        kfr.u[1] = *(const uint4*)(rp + 16 + sel * 8);
        sacc[ns] = __builtin_amdgcn_wmma_f32_16x16x32_bf16(
            false, qa[kf].v, false, kfr.v, (short)0, sacc[ns], false, false);
      }
    }

    // ---- scale + ALiBi + causal, online softmax (base-2 domain) --------
    float tl[2][8], tm[8];
#pragma unroll
    for (int e = 0; e < 8; ++e) {
      int qidx = qrow0 + sel * 8 + e;
      float best = -__builtin_inff();
#pragma unroll
      for (int ns = 0; ns < 2; ++ns) {
        int kidx = kb + ns * 16 + r;
        float tv = sacc[ns][e] * SCL + slope2 * (float)(qidx - kidx);
        if (kidx > qidx) tv = -__builtin_inff();
        tl[ns][e] = tv;
        best = fmaxf(best, tv);
      }
      tm[e] = best;
    }
#pragma unroll
    for (int off = 8; off >= 1; off >>= 1)
#pragma unroll
      for (int e = 0; e < 8; ++e)
        tm[e] = fmaxf(tm[e], __shfl_xor(tm[e], off, 32));  // 16-lane row reduce

    float corr[8], rs[8];
#pragma unroll
    for (int e = 0; e < 8; ++e) {
      float nm = fmaxf(mrun[e], tm[e]);
      corr[e]  = exp2f(mrun[e] - nm);
      mrun[e]  = nm;
      float p0 = exp2f(tl[0][e] - nm);
      float p1 = exp2f(tl[1][e] - nm);
      tl[0][e] = p0; tl[1][e] = p1;
      rs[e] = p0 + p1;
    }
#pragma unroll
    for (int off = 8; off >= 1; off >>= 1)
#pragma unroll
      for (int e = 0; e < 8; ++e) rs[e] += __shfl_xor(rs[e], off, 32);
#pragma unroll
    for (int e = 0; e < 8; ++e) lrun[e] = lrun[e] * corr[e] + rs[e];
#pragma unroll
    for (int i = 0; i < 4; ++i)
#pragma unroll
      for (int e = 0; e < 8; ++e) oacc[i][e] *= corr[e];

    // ---- transpose P (C-layout -> A-layout) via per-wave LDS -----------
    __bf16* pw = &lP[wave * 16 * 40];
#pragma unroll
    for (int ns = 0; ns < 2; ++ns)
#pragma unroll
      for (int e = 0; e < 8; ++e)
        pw[(sel * 8 + e) * 40 + ns * 16 + r] = f2bf(tl[ns][e]);
    __syncthreads();

    // ---- O += P . V ----------------------------------------------------
    Frag16 pf;
    {
      const __bf16* rp = &pw[r * 40];
      pf.u[0] = *(const uint4*)(rp + sel * 8);
      pf.u[1] = *(const uint4*)(rp + 16 + sel * 8);
    }
#pragma unroll
    for (int dsu = 0; dsu < 4; ++dsu) {
      Frag16 vf;
      const __bf16* rp = &lVt[(dsu * 16 + r) * 40];
      vf.u[0] = *(const uint4*)(rp + sel * 8);
      vf.u[1] = *(const uint4*)(rp + 16 + sel * 8);
      oacc[dsu] = __builtin_amdgcn_wmma_f32_16x16x32_bf16(
          false, pf.v, false, vf.v, (short)0, oacc[dsu], false, false);
    }
    __syncthreads();   // protect lK/lVt before next-iteration staging
  }

  // ---- normalize and store O (bf16, [B,H,S,D] contiguous) --------------
  float inv[8];
#pragma unroll
  for (int e = 0; e < 8; ++e) inv[e] = 1.0f / lrun[e];
  __bf16* Ob = O + (long)bh * S * DHEAD;
#pragma unroll
  for (int dsu = 0; dsu < 4; ++dsu)
#pragma unroll
    for (int e = 0; e < 8; ++e) {
      int qidx = qrow0 + sel * 8 + e;
      int d    = dsu * 16 + r;
      Ob[(long)qidx * DHEAD + d] = f2bf(oacc[dsu][e] * inv[e]);
    }
}

// ---------------------------------------------------------------------------
extern "C" void kernel_launch(void* const* d_in, const int* in_sizes, int n_in,
                              void* d_out, int out_size, void* d_ws, size_t ws_size,
                              hipStream_t stream) {
  const float* x  = (const float*)d_in[0];
  const float* Wq = (const float*)d_in[1];
  const float* bq = (const float*)d_in[2];
  const float* Wk = (const float*)d_in[3];
  const float* bk = (const float*)d_in[4];
  const float* Wv = (const float*)d_in[5];
  const float* bv = (const float*)d_in[6];
  const float* Wo = (const float*)d_in[7];
  const float* bo = (const float*)d_in[8];

  const int B = 4, S = 2048, E = 768;
  const long nx   = (long)B * S * E;                 // 6,291,456
  const long nw   = (long)E * E;                     //   589,824
  const long nqkv = (long)B * H_HEADS * S * DHEAD;   // == nx

  char*   ws  = (char*)d_ws;
  __bf16* xb  = (__bf16*)(ws);                        // 12.6 MB
  __bf16* wqt = (__bf16*)(ws + nx * 2);               // 1.18 MB each
  __bf16* wkt = (__bf16*)(ws + nx * 2 + nw * 2);
  __bf16* wvt = (__bf16*)(ws + nx * 2 + nw * 4);
  __bf16* wot = (__bf16*)(ws + nx * 2 + nw * 6);
  __bf16* q   = (__bf16*)(ws + nx * 2 + nw * 8);      // 12.6 MB each
  __bf16* k   = (__bf16*)(ws + nx * 2 + nw * 8 + nqkv * 2);
  __bf16* v   = (__bf16*)(ws + nx * 2 + nw * 8 + nqkv * 4);
  __bf16* o   = (__bf16*)(ws + nx * 2 + nw * 8 + nqkv * 6);

  // one-time precision / layout conversion
  cvt_x<<<dim3((unsigned)(nx / 8 / 256)), 256, 0, stream>>>(x, xb, nx);
  dim3 gT(E / 64, E / 64);                           // (12,12)
  transpose_w<<<gT, 256, 0, stream>>>(Wq, wqt, E, E);
  transpose_w<<<gT, 256, 0, stream>>>(Wk, wkt, E, E);
  transpose_w<<<gT, 256, 0, stream>>>(Wv, wvt, E, E);
  transpose_w<<<gT, 256, 0, stream>>>(Wo, wot, E, E);

  dim3 gGemm(E / BN, (B * S) / BM);                  // (6, 64)
  gemm_kernel<<<gGemm, 256, 0, stream>>>(xb, wqt, bq, nullptr, q, B * S, E, E, S, 1);
  gemm_kernel<<<gGemm, 256, 0, stream>>>(xb, wkt, bk, nullptr, k, B * S, E, E, S, 1);
  gemm_kernel<<<gGemm, 256, 0, stream>>>(xb, wvt, bv, nullptr, v, B * S, E, E, S, 1);

  attn_kernel<<<dim3(S / 64, B * H_HEADS), 128, 0, stream>>>(q, k, v, o, S);

  // O ([B,H,S,D] contiguous) reinterpreted as [B*S, E] reproduces the
  // reference's head-major reshape; final projection writes fp32 d_out.
  gemm_kernel<<<gGemm, 256, 0, stream>>>(o, wot, bo, (float*)d_out, nullptr,
                                         B * S, E, E, S, 0);
}
